// InfoNCELoss_12395275616921
// MI455X (gfx1250) — matrix operand order repaired and checked by reading, not compiled, not run on hardware
//
#include <hip/hip_runtime.h>
#include <math.h>

typedef __attribute__((ext_vector_type(16))) _Float16 v16h;
typedef __attribute__((ext_vector_type(8)))  _Float16 v8h;
typedef __attribute__((ext_vector_type(8)))  float    v8f;

// RHO = 10, TAU = 0.1  ->  off-diagonal raw = 10*S ; fixed shift of 10:
//   raw' = 10*S - 10  in [-20, 0]
// loss = sum_{pos (i,j)} [ log(exp(raw'_ij) + neg_i) - raw'_ij ] / num_pos
// neg_i = sum_{j: class differs} exp(raw'_ij)

// ---------------------------------------------------------------------------
// Kernel 1: L2-normalize each row, emit f16 copy for WMMA.
// ---------------------------------------------------------------------------
__global__ __launch_bounds__(256)
void infonce_normalize(const float* __restrict__ emb, _Float16* __restrict__ nh, int Dd) {
    const int row = blockIdx.x;
    const float* e = emb + (size_t)row * Dd;

    float ss = 0.0f;
    for (int t = threadIdx.x; t < Dd; t += 256) {
        float v = e[t];
        ss += v * v;
    }
    // wave32 reduction
    ss += __shfl_xor(ss, 16, 32);
    ss += __shfl_xor(ss,  8, 32);
    ss += __shfl_xor(ss,  4, 32);
    ss += __shfl_xor(ss,  2, 32);
    ss += __shfl_xor(ss,  1, 32);

    __shared__ float sw[8];
    __shared__ float stot;
    if ((threadIdx.x & 31) == 0) sw[threadIdx.x >> 5] = ss;
    __syncthreads();
    if (threadIdx.x == 0) {
        float t = 0.0f;
        for (int w = 0; w < 8; ++w) t += sw[w];
        stot = t;
    }
    __syncthreads();

    const float inv = 1.0f / fmaxf(sqrtf(stot), 1e-12f);
    _Float16* o = nh + (size_t)row * Dd;
    for (int t = threadIdx.x; t < Dd; t += 256) {
        o[t] = (_Float16)(e[t] * inv);
    }
}

// ---------------------------------------------------------------------------
// GEMM passes. Block = 256 threads = 8 waves, owns 64 full rows of S and
// loops over all columns in 128-wide strips. Wave (rg,cg): rows rg*16..+16,
// cols cg*64..+64 within the strip; 4 accumulators (16x64 per wave).
//
// PASS 0: neg_i (per-row, written once, deterministically), block partials of
//         sum(raw'_pos) and count(pos).
// PASS 1: block partials of sum(log(exp(raw') + neg_i)) over positives.
// ---------------------------------------------------------------------------
template <int PASS>
__global__ __launch_bounds__(256)
void infonce_gemm(const _Float16* __restrict__ nh,
                  const int*      __restrict__ cls,
                  float*          __restrict__ neg,
                  float*          __restrict__ part0,   // PASS0: sum raw'  | PASS1: sum log
                  float*          __restrict__ part1,   // PASS0: count    | PASS1: unused
                  int N, int Dd) {
    __shared__ float negL[2][64];
    __shared__ float sA[8], sB[8];

    const int tid  = threadIdx.x;
    const int lane = tid & 31;
    const int wid  = tid >> 5;
    const int rg   = wid >> 1;      // 0..3 row group
    const int cg   = wid & 1;       // 0..1 col group
    const int hi   = lane >> 4;     // half-wave
    const int nloc = lane & 15;

    const int rowW   = blockIdx.x * 64 + rg * 16;  // wave's 16-row base
    const int myRow0 = rowW + hi * 8;              // C-frag rows myRow0 + r

    // A fragment base for this lane (M = nloc), K split per half-wave (+0 / +8)
    const _Float16* aBase = nh + (size_t)(rowW + nloc) * Dd + hi * 8;

    int   myCls[8];
    float myNeg[8];
#pragma unroll
    for (int r = 0; r < 8; ++r) {
        myCls[r] = cls[myRow0 + r];
        myNeg[r] = (PASS == 1) ? neg[myRow0 + r] : 0.0f;
    }

    float negAcc[8] = {0, 0, 0, 0, 0, 0, 0, 0};
    float rawAcc = 0.0f, cntAcc = 0.0f, logAcc = 0.0f;

    const int KT = Dd >> 5;   // K tiles of 32
    const int CT = N >> 7;    // column strips of 128

    for (int ct = 0; ct < CT; ++ct) {
        const int jb = (ct << 7) + (cg << 6);

        int jlane[4];
        int clsJ[4];
        const _Float16* bBase[4];
#pragma unroll
        for (int cc = 0; cc < 4; ++cc) {
            jlane[cc] = jb + (cc << 4) + nloc;
            clsJ[cc]  = cls[jlane[cc]];
            bBase[cc] = nh + (size_t)jlane[cc] * Dd + hi * 8;
            if (jlane[cc] + 128 < N)  // prefetch next strip's B rows (global_prefetch_b8)
                __builtin_prefetch(nh + (size_t)(jlane[cc] + 128) * Dd, 0, 1);
        }

        v8f acc[4];
#pragma unroll
        for (int cc = 0; cc < 4; ++cc) acc[cc] = (v8f){0, 0, 0, 0, 0, 0, 0, 0};

        for (int kt = 0; kt < KT; ++kt) {
            const int k0 = kt << 5;
            // A fragment: lanes 0-15 K {0..7,16..23}, lanes 16-31 K {8..15,24..31}
            v8h alo = *(const v8h*)(aBase + k0);
            v8h ahi = *(const v8h*)(aBase + k0 + 16);
            v16h av = __builtin_shufflevector(alo, ahi,
                         0, 1, 2, 3, 4, 5, 6, 7, 8, 9, 10, 11, 12, 13, 14, 15);
#pragma unroll
            for (int cc = 0; cc < 4; ++cc) {
                // B fragment: lanes 0-15 K {0..7,16..23}, lanes 16-31 K {8..15,24..31}
                v8h blo = *(const v8h*)(bBase[cc] + k0);
                v8h bhi = *(const v8h*)(bBase[cc] + k0 + 16);
                v16h bv = __builtin_shufflevector(blo, bhi,
                             0, 1, 2, 3, 4, 5, 6, 7, 8, 9, 10, 11, 12, 13, 14, 15);
                acc[cc] = __builtin_amdgcn_wmma_f32_16x16x32_f16(
                    false, av, false, bv, (short)0, acc[cc], false, false);
            }
        }

        // fused epilogue: D element (M = myRow0 + r, N = jlane[cc])
#pragma unroll
        for (int cc = 0; cc < 4; ++cc) {
#pragma unroll
            for (int r = 0; r < 8; ++r) {
                const int   i    = myRow0 + r;
                const float rawp = acc[cc][r] * 10.0f - 10.0f;
                const bool  same = (myCls[r] == clsJ[cc]);
                if (PASS == 0) {
                    if (!same) {
                        negAcc[r] += __expf(rawp);            // different-class
                    } else if (i != jlane[cc]) {
                        rawAcc += rawp;                        // positive pair
                        cntAcc += 1.0f;
                    }
                } else {
                    if (same && i != jlane[cc])
                        logAcc += __logf(__expf(rawp) + myNeg[r]);
                }
            }
        }
    }

    // ---- per-row neg: reduce across 16 lanes of each half-wave (fixed order)
    if (PASS == 0) {
#pragma unroll
        for (int r = 0; r < 8; ++r) {
            float v = negAcc[r];
            v += __shfl_xor(v, 1, 32);
            v += __shfl_xor(v, 2, 32);
            v += __shfl_xor(v, 4, 32);
            v += __shfl_xor(v, 8, 32);
            negAcc[r] = v;
        }
        if (nloc == 0) {
#pragma unroll
            for (int r = 0; r < 8; ++r)
                negL[cg][rg * 16 + hi * 8 + r] = negAcc[r];
        }
    }

    // ---- scalar partials: full-wave reduce, then fixed-order LDS combine
    float s0 = (PASS == 0) ? rawAcc : logAcc;
    float s1 = cntAcc;
    s0 += __shfl_xor(s0, 16, 32); s1 += __shfl_xor(s1, 16, 32);
    s0 += __shfl_xor(s0,  8, 32); s1 += __shfl_xor(s1,  8, 32);
    s0 += __shfl_xor(s0,  4, 32); s1 += __shfl_xor(s1,  4, 32);
    s0 += __shfl_xor(s0,  2, 32); s1 += __shfl_xor(s1,  2, 32);
    s0 += __shfl_xor(s0,  1, 32); s1 += __shfl_xor(s1,  1, 32);
    if (lane == 0) { sA[wid] = s0; sB[wid] = s1; }
    __syncthreads();

    if (PASS == 0 && tid < 64)
        neg[blockIdx.x * 64 + tid] = negL[0][tid] + negL[1][tid];

    if (tid == 0) {
        float t0 = 0.0f, t1 = 0.0f;
        for (int w = 0; w < 8; ++w) { t0 += sA[w]; t1 += sB[w]; }
        part0[blockIdx.x] = t0;
        if (PASS == 0) part1[blockIdx.x] = t1;
    }
}

// ---------------------------------------------------------------------------
// Finalize: deterministic fixed-order reduction of per-block partials.
// ---------------------------------------------------------------------------
__global__ void infonce_finalize(const float* __restrict__ rawPart,
                                 const float* __restrict__ cntPart,
                                 const float* __restrict__ logPart,
                                 float* __restrict__ out, int nb) {
    if (blockIdx.x == 0 && threadIdx.x == 0) {
        float raw = 0.0f, cnt = 0.0f, lg = 0.0f;
        for (int b = 0; b < nb; ++b) {
            raw += rawPart[b];
            cnt += cntPart[b];
            lg  += logPart[b];
        }
        const float total = lg - raw;
        out[0] = (cnt > 0.0f) ? (total / cnt) : 0.0f;
    }
}

// ---------------------------------------------------------------------------
extern "C" void kernel_launch(void* const* d_in, const int* in_sizes, int n_in,
                              void* d_out, int out_size, void* d_ws, size_t ws_size,
                              hipStream_t stream) {
    const float* emb = (const float*)d_in[0];
    const int*   cls = (const int*)d_in[1];
    float*       out = (float*)d_out;

    const int N  = in_sizes[1];
    const int Dd = in_sizes[0] / N;          // 512
    const int NB = N / 64;                   // 128 GEMM blocks

    char* w = (char*)d_ws;
    _Float16* nh = (_Float16*)w;
    const size_t nhB = (size_t)N * Dd * sizeof(_Float16);
    float* neg     = (float*)(w + nhB);      // [N]
    float* rawPart = neg + N;                // [NB]
    float* cntPart = rawPart + NB;           // [NB]
    float* logPart = cntPart + NB;           // [NB]

    infonce_normalize<<<N, 256, 0, stream>>>(emb, nh, Dd);
    infonce_gemm<0><<<NB, 256, 0, stream>>>(nh, cls, neg, rawPart, cntPart, N, Dd);
    infonce_gemm<1><<<NB, 256, 0, stream>>>(nh, cls, neg, logPart, nullptr, N, Dd);
    infonce_finalize<<<1, 32, 0, stream>>>(rawPart, cntPart, logPart, out, NB);
}
